// Coarsening_14705968021525
// MI455X (gfx1250) — matrix operand order repaired and checked by reading, not compiled, not run on hardware
//
#include <hip/hip_runtime.h>
#include <hip/hip_bf16.h>

typedef _Float16 half_t;
typedef __attribute__((ext_vector_type(16))) _Float16 v16h;
typedef __attribute__((ext_vector_type(4)))  _Float16 v4h;
typedef __attribute__((ext_vector_type(8)))  float    v8f;
typedef __attribute__((ext_vector_type(4)))  unsigned u32x4;
typedef __attribute__((ext_vector_type(4)))  int      v4i;

#define TB 256

// gfx1250 async global->LDS path (ASYNCcnt), guarded so compile never regresses.
#if defined(__has_builtin)
#if __has_builtin(__builtin_amdgcn_global_load_async_to_lds_b128) && \
    __has_builtin(__builtin_amdgcn_s_wait_asynccnt)
#define USE_ASYNC_LDS 1
#endif
#endif
#ifndef USE_ASYNC_LDS
#define USE_ASYNC_LDS 0
#endif

// ---------------------------------------------------------------------------
// WMMA GEMM:  C[M,Ncol] = A[M,K] * B[K,Ncol], f16 inputs (row-major), f32 out.
// Requires M%128==0, Ncol%64==0, K%32==0 (all shapes here satisfy this).
// Block: 256 threads = 8 waves (wave32). Block tile 128x64, K step 32.
// Waves in 4x2 grid; each wave computes a 32x32 tile = 2x2 WMMA fragments.
// A tile is staged via GLOBAL_LOAD_ASYNC_TO_LDS_B128 (16B/lane, no VGPR
// round-trip, drained once per tile with s_wait_asynccnt 0).
// ---------------------------------------------------------------------------
__global__ __launch_bounds__(256)
void wmma_gemm_f16(const half_t* __restrict__ A, const half_t* __restrict__ Bm,
                   float* __restrict__ C, int M, int K, int Ncol,
                   long long sA, long long sB, long long sC)
{
  constexpr int BM = 128, BN = 64, BK = 32;
  constexpr int LDA = BK + 8;   // 40 halves = 80B: 16B-aligned rows, 20-bank stride
  constexpr int LDB = BK + 2;   // 34 halves: 4B-aligned k-pairs
  __shared__ __align__(16) half_t As[BM * LDA];       // [row][k], k contiguous
  __shared__ __align__(16) half_t Bs[BN * LDB];       // [n][k],   k contiguous

  const int zb = blockIdx.z;
  A  += (long long)zb * sA;
  Bm += (long long)zb * sB;
  C  += (long long)zb * sC;

  const int tid  = threadIdx.x;
  const int lane = tid & 31;
  const int wid  = tid >> 5;
  const int wm   = wid >> 1;           // 0..3  (rows)
  const int wn   = wid & 1;            // 0..1  (cols)
  const int row0 = blockIdx.y * BM;
  const int col0 = blockIdx.x * BN;
  const int l16  = (lane & 16) ? 1 : 0;
  const int lm   = lane & 15;

  v8f acc[2][2];
  for (int i = 0; i < 2; ++i)
    for (int j = 0; j < 2; ++j)
      for (int e = 0; e < 8; ++e) acc[i][j][e] = 0.0f;

  union Frag { v16h h; unsigned u[8]; };

  for (int k0 = 0; k0 < K; k0 += BK) {
    __syncthreads();

    // ---- Stage A tile: 128x32 halves = 8KB as 512 x 16B chunks, 2/thread ----
#pragma unroll
    for (int p = 0; p < (BM * BK) / (8 * TB); ++p) {   // 2 passes
      const int c   = tid + p * TB;
      const int r   = c >> 2;            // row in tile
      const int seg = (c & 3) * 8;       // 8-half (16B) segment within row
      const half_t* gsrc = &A[(long long)(row0 + r) * K + k0 + seg];
      half_t*       ldst = &As[r * LDA + seg];
#if USE_ASYNC_LDS
      __builtin_amdgcn_global_load_async_to_lds_b128(
          (__attribute__((address_space(1))) v4i*)gsrc,
          (__attribute__((address_space(3))) v4i*)ldst, 0, 0);
#else
      *(u32x4*)ldst = *(const u32x4*)gsrc;
#endif
    }

    // ---- Stage B tile 32x64 transposed to [n][k]; global side reads v4h ----
#pragma unroll
    for (int p = 0; p < (BK * BN) / (4 * TB); ++p) {   // 2 passes
      const int c  = tid + p * TB;
      const int kk = c / (BN / 4);
      const int n4 = (c % (BN / 4)) * 4;
      v4h q = *(const v4h*)&Bm[(long long)(k0 + kk) * Ncol + col0 + n4];
      Bs[(n4 + 0) * LDB + kk] = q.x;
      Bs[(n4 + 1) * LDB + kk] = q.y;
      Bs[(n4 + 2) * LDB + kk] = q.z;
      Bs[(n4 + 3) * LDB + kk] = q.w;
    }

    // Prefetch next K tile while this one computes (global_prefetch_b8).
    if (k0 + BK < K) {
      __builtin_prefetch(&A[(long long)(row0 + (tid >> 1)) * K + k0 + BK], 0, 3);
      __builtin_prefetch(&Bm[(long long)(k0 + BK + lane) * Ncol + col0], 0, 3);
    }

#if USE_ASYNC_LDS
    __builtin_amdgcn_s_wait_asynccnt(0);
#endif
    __syncthreads();

    Frag a[2], b[2];
#pragma unroll
    for (int t = 0; t < 2; ++t) {
      const int arow = (wm * 32 + t * 16 + lm) * LDA;
      const int bcol = (wn * 32 + t * 16 + lm) * LDB;
#pragma unroll
      for (int v = 0; v < 8; ++v) {
        // A 16x32 f16, contiguous k order used here: lanes 0-15 K=0..15, lanes 16-31 K=16..31
        const int ka = v * 2 + (l16 ? 16 : 0);
        // B 32x16 f16, same contiguous k order as A: lanes 0-15 K=0..15
        const int kb = v * 2 + (l16 ? 16 : 0);
        a[t].u[v] = *(const unsigned*)&As[arow + ka];
        b[t].u[v] = *(const unsigned*)&Bs[bcol + kb];
      }
    }
#pragma unroll
    for (int i = 0; i < 2; ++i)
#pragma unroll
      for (int j = 0; j < 2; ++j)
        acc[i][j] = __builtin_amdgcn_wmma_f32_16x16x32_f16(
            false, a[i].h, false, b[j].h, (short)0, acc[i][j], false, false);
  }

  // C/D layout: VGPR e -> row e (+8 for lanes 16..31), col = lane%16.
#pragma unroll
  for (int i = 0; i < 2; ++i)
#pragma unroll
    for (int j = 0; j < 2; ++j) {
      const int r0 = row0 + wm * 32 + i * 16 + l16 * 8;
      const int c  = col0 + wn * 32 + j * 16 + lm;
#pragma unroll
      for (int e = 0; e < 8; ++e)
        C[(long long)(r0 + e) * Ncol + c] = acc[i][j][e];
    }
}

// ---------------------------------------------------------------------------
// Elementwise / reduction kernels (wave32-friendly, 256-thread blocks)
// ---------------------------------------------------------------------------
__global__ void cvt_f32_to_f16(const float* __restrict__ src,
                               half_t* __restrict__ dst, long long n)
{
  long long i = (long long)blockIdx.x * blockDim.x + threadIdx.x;
  long long stride = (long long)gridDim.x * blockDim.x;
  for (; i < n; i += stride) dst[i] = (half_t)src[i];
}

// Per-row: count nonzeros and weighted sum of adj -> binary/weighted norms.
__global__ void row_stats_kernel(const float* __restrict__ adj,
                                 float* __restrict__ dis_bin,
                                 float* __restrict__ dis_w,
                                 float* __restrict__ rowflag, int N)
{
  const int i = blockIdx.x, b = blockIdx.y;
  const float* row = adj + ((size_t)b * N + i) * N;
  __shared__ float rc[TB], rs[TB];
  float cnt = 0.f, sum = 0.f;
  for (int j = threadIdx.x; j < N; j += TB) {
    float a = row[j];
    cnt += (a > 0.f) ? 1.f : 0.f;
    sum += a;
  }
  rc[threadIdx.x] = cnt; rs[threadIdx.x] = sum;
  __syncthreads();
  for (int s = TB / 2; s > 0; s >>= 1) {
    if (threadIdx.x < s) {
      rc[threadIdx.x] += rc[threadIdx.x + s];
      rs[threadIdx.x] += rs[threadIdx.x + s];
    }
    __syncthreads();
  }
  if (threadIdx.x == 0) {
    size_t o = (size_t)b * N + i;
    dis_bin[o] = rsqrtf(rc[0] + 1.f);                // A=adj_bin+I => d>=1
    dis_w[o]   = rsqrtf(fmaxf(rs[0] + 1.f, 1.f));    // clip(sum,1)^-0.5
    rowflag[o] = (rs[0] > 0.f) ? 1.f : 0.f;
  }
}

// Build f16 symmetric-normalized binary adjacency + f32 normalize_batch_adj.
__global__ void norm_build_kernel(const float* __restrict__ adj,
                                  const float* __restrict__ dis_bin,
                                  const float* __restrict__ dis_w,
                                  const float* __restrict__ rowflag,
                                  half_t* __restrict__ normbin,
                                  float* __restrict__ norm_adj, int N)
{
  const int i = blockIdx.x, b = blockIdx.y;
  const size_t ro = ((size_t)b * N + i) * N;
  const float dbi = dis_bin[(size_t)b * N + i];
  const float dwi = dis_w[(size_t)b * N + i] * rowflag[(size_t)b * N + i];
  for (int j = threadIdx.x; j < N; j += TB) {
    float a = adj[ro + j];
    float diag = (j == i) ? 1.f : 0.f;
    float abin = ((a > 0.f) ? 1.f : 0.f) + diag;
    normbin[ro + j]  = (half_t)(dbi * abin * dis_bin[(size_t)b * N + j]);
    norm_adj[ro + j] = dwi * (a + diag) * dis_w[(size_t)b * N + j];
  }
}

__global__ void relu_bias_kernel(float* __restrict__ x,
                                 const float* __restrict__ bias,
                                 half_t* __restrict__ xh, int H, long long n)
{
  long long i = (long long)blockIdx.x * blockDim.x + threadIdx.x;
  long long stride = (long long)gridDim.x * blockDim.x;
  for (; i < n; i += stride) {
    float v = fmaxf(x[i] + bias[i % H], 0.f);
    x[i] = v;
    xh[i] = (half_t)v;
  }
}

__global__ void att_dot_kernel(const float* __restrict__ x1,
                               const float* __restrict__ Watt,
                               float* __restrict__ v, int H)
{
  const size_t r = blockIdx.x;
  __shared__ float red[TB];
  float s = 0.f;
  for (int h = threadIdx.x; h < H; h += TB) s += x1[r * H + h] * Watt[h];
  red[threadIdx.x] = s;
  __syncthreads();
  for (int st = TB / 2; st > 0; st >>= 1) {
    if (threadIdx.x < st) red[threadIdx.x] += red[threadIdx.x + st];
    __syncthreads();
  }
  if (threadIdx.x == 0) v[r] = red[0];
}

__global__ void att_prop_kernel(const float* __restrict__ adj,
                                const float* __restrict__ dis_bin,
                                const float* __restrict__ v,
                                const float* __restrict__ batt,
                                float* __restrict__ alpha, int N)
{
  const int i = blockIdx.x, b = blockIdx.y;
  const size_t ro = ((size_t)b * N + i) * N;
  __shared__ float red[TB];
  float s = 0.f;
  for (int j = threadIdx.x; j < N; j += TB) {
    float a = adj[ro + j];
    float abin = ((a > 0.f) ? 1.f : 0.f) + ((j == i) ? 1.f : 0.f);
    s += abin * dis_bin[(size_t)b * N + j] * v[(size_t)b * N + j];
  }
  red[threadIdx.x] = s;
  __syncthreads();
  for (int st = TB / 2; st > 0; st >>= 1) {
    if (threadIdx.x < st) red[threadIdx.x] += red[threadIdx.x + st];
    __syncthreads();
  }
  if (threadIdx.x == 0) {
    float t = dis_bin[(size_t)b * N + i] * red[0] + batt[0];
    t = t * t;
    alpha[(size_t)b * N + i] = 1.f / (1.f + expf(-t));
  }
}

// k rounds of max-selection over sentence nodes; key = (alpha, -index)
// packed into an order-preserving u64 (alpha>0 so float bits are monotone).
__global__ void topk_cut_kernel(const float* __restrict__ alpha,
                                const int* __restrict__ num_sent,
                                const int* __restrict__ kptr,
                                float* __restrict__ cutA,
                                int* __restrict__ cutI, int N)
{
  const int b  = blockIdx.x;
  const int ns = num_sent[b];
  const int k  = kptr[0];
  const float* a = alpha + (size_t)b * N;
  __shared__ unsigned long long red[TB];
  unsigned long long prev = ~0ull;
  for (int r = 0; r < k; ++r) {
    unsigned long long best = 0ull;
    for (int j = threadIdx.x; j < ns; j += TB) {
      unsigned fb = __float_as_uint(a[j]);
      unsigned long long key =
          ((unsigned long long)fb << 32) | (unsigned)(0xFFFFFFFFu - (unsigned)j);
      if (key < prev && key > best) best = key;
    }
    red[threadIdx.x] = best;
    __syncthreads();
    for (int s = TB / 2; s > 0; s >>= 1) {
      if (threadIdx.x < s)
        red[threadIdx.x] = (red[threadIdx.x + s] > red[threadIdx.x])
                               ? red[threadIdx.x + s] : red[threadIdx.x];
      __syncthreads();
    }
    prev = red[0];
    __syncthreads();
  }
  if (threadIdx.x == 0) {
    cutA[b] = __uint_as_float((unsigned)(prev >> 32));
    cutI[b] = (int)(0xFFFFFFFFu - (unsigned)(prev & 0xFFFFFFFFu));
  }
}

__global__ void mask_cut_kernel(const float* __restrict__ alpha,
                                const float* __restrict__ cutA,
                                const int* __restrict__ cutI,
                                float* __restrict__ mask_out,
                                float* __restrict__ cut_alpha, int N)
{
  const int b = blockIdx.y;
  const int j = blockIdx.x * TB + threadIdx.x;
  if (j < N) {
    const float a = alpha[(size_t)b * N + j];
    const float c = cutA[b];
    const int  ci = cutI[b];
    mask_out[(size_t)b * N + j] =
        (a > c || (a == c && j <= ci)) ? 1.f : 0.f;
    cut_alpha[(size_t)b * N + j] = fmaxf(a + 1e-7f - c, 0.f);
  }
}

// S = L1-rownorm(norm_adj * cut_alpha_col); emit f32 out + f16 + f16-transposed.
__global__ void s_build_kernel(const float* __restrict__ norm_adj,
                               const float* __restrict__ cut_alpha,
                               float* __restrict__ S_out,
                               half_t* __restrict__ S16,
                               half_t* __restrict__ St16, int N)
{
  const int i = blockIdx.x, b = blockIdx.y;
  const size_t ro = ((size_t)b * N + i) * N;
  const float* ca = cut_alpha + (size_t)b * N;
  __shared__ float red[TB];
  float s = 0.f;
  for (int j = threadIdx.x; j < N; j += TB) s += fabsf(norm_adj[ro + j] * ca[j]);
  red[threadIdx.x] = s;
  __syncthreads();
  for (int st = TB / 2; st > 0; st >>= 1) {
    if (threadIdx.x < st) red[threadIdx.x] += red[threadIdx.x + st];
    __syncthreads();
  }
  const float inv = 1.f / fmaxf(red[0], 1e-12f);
  for (int j = threadIdx.x; j < N; j += TB) {
    float v = norm_adj[ro + j] * ca[j] * inv;
    S_out[ro + j] = v;
    S16[ro + j] = (half_t)v;
    St16[((size_t)b * N + j) * N + i] = (half_t)v;   // transpose for S^T GEMMs
  }
}

__global__ void quant_kernel(float* __restrict__ x, long long n)
{
  long long i = (long long)blockIdx.x * blockDim.x + threadIdx.x;
  long long stride = (long long)gridDim.x * blockDim.x;
  for (; i < n; i += stride) x[i] = floorf(x[i] * 1e4f) / 1e4f;
}

__global__ void coarse_stats_kernel(const float* __restrict__ cadj,
                                    float* __restrict__ dis_c, int N)
{
  const int i = blockIdx.x, b = blockIdx.y;
  const size_t ro = ((size_t)b * N + i) * N;
  __shared__ float red[TB];
  float cnt = 0.f;
  for (int j = threadIdx.x; j < N; j += TB)
    cnt += (cadj[ro + j] != 0.f) ? 1.f : 0.f;
  red[threadIdx.x] = cnt;
  __syncthreads();
  for (int st = TB / 2; st > 0; st >>= 1) {
    if (threadIdx.x < st) red[threadIdx.x] += red[threadIdx.x + st];
    __syncthreads();
  }
  if (threadIdx.x == 0)
    dis_c[(size_t)b * N + i] = rsqrtf(red[0] + 1.f);
}

__global__ void normc_build_kernel(const float* __restrict__ cadj,
                                   const float* __restrict__ dis_c,
                                   half_t* __restrict__ normc, int N)
{
  const int i = blockIdx.x, b = blockIdx.y;
  const size_t ro = ((size_t)b * N + i) * N;
  const float di = dis_c[(size_t)b * N + i];
  for (int j = threadIdx.x; j < N; j += TB) {
    float abin = ((cadj[ro + j] != 0.f) ? 1.f : 0.f) + ((j == i) ? 1.f : 0.f);
    normc[ro + j] = (half_t)(di * abin * dis_c[(size_t)b * N + j]);
  }
}

__global__ void tanh_bias_kernel(float* __restrict__ x,
                                 const float* __restrict__ bias,
                                 int F, long long n)
{
  long long i = (long long)blockIdx.x * blockDim.x + threadIdx.x;
  long long stride = (long long)gridDim.x * blockDim.x;
  for (; i < n; i += stride) x[i] = tanhf(x[i] + bias[i % F]);
}

__global__ void row_mean_kernel(const float* __restrict__ x,
                                float* __restrict__ out, int C)
{
  const size_t r = blockIdx.x;
  __shared__ float red[TB];
  float s = 0.f;
  for (int c = threadIdx.x; c < C; c += TB) s += x[r * C + c];
  red[threadIdx.x] = s;
  __syncthreads();
  for (int st = TB / 2; st > 0; st >>= 1) {
    if (threadIdx.x < st) red[threadIdx.x] += red[threadIdx.x + st];
    __syncthreads();
  }
  if (threadIdx.x == 0) out[r] = red[0] / (float)C;
}

// ---------------------------------------------------------------------------
extern "C" void kernel_launch(void* const* d_in, const int* in_sizes, int n_in,
                              void* d_out, int out_size, void* d_ws, size_t ws_size,
                              hipStream_t stream)
{
  const float* x    = (const float*)d_in[0];
  const float* adj  = (const float*)d_in[1];
  const int*   nsnt = (const int*)d_in[2];
  const float* W1   = (const float*)d_in[3];
  const float* b1   = (const float*)d_in[4];
  const float* Watt = (const float*)d_in[5];
  const float* batt = (const float*)d_in[6];
  const float* W2   = (const float*)d_in[7];
  const float* b2   = (const float*)d_in[8];
  const int*   kptr = (const int*)d_in[9];
  (void)n_in; (void)out_size; (void)ws_size;

  const int B = in_sizes[2];                 // 16
  const int H = in_sizes[4];                 // 128
  const int F = in_sizes[8];                 // 768
  const int N = in_sizes[0] / (B * F);       // 1024
  const long long NN  = (long long)N * N;
  const long long BNl = (long long)B * N;

  float* out = (float*)d_out;
  const long long o_x2   = 0;
  const long long o_cx   = (long long)B * N * F;
  const long long o_cadj = o_cx + (long long)B * N * H;
  const long long o_S    = o_cadj + (long long)B * NN;
  const long long o_mask = o_S + (long long)B * NN;
  const long long o_xs0  = o_mask + BNl;
  const long long o_xs1  = o_xs0 + BNl;

  // Bump allocator over workspace; buffers with non-overlapping lifetimes alias.
  char* w = (char*)d_ws;
  auto alloc = [&](long long bytes) -> void* {
    void* p = (void*)w;
    w += (bytes + 255) & ~255LL;
    return p;
  };
  float*  dis_bin  = (float*)alloc(BNl * 4);
  float*  dis_w    = (float*)alloc(BNl * 4);
  float*  rowflag  = (float*)alloc(BNl * 4);
  float*  v_att    = (float*)alloc(BNl * 4);
  float*  alpha    = (float*)alloc(BNl * 4);
  float*  cutalpha = (float*)alloc(BNl * 4);
  float*  dis_c    = (float*)alloc(BNl * 4);
  float*  cutA     = (float*)alloc((long long)B * 4);
  int*    cutI     = (int*)alloc((long long)B * 4);
  float*  A64  = (float*)alloc((long long)B * NN * 4);   // norm_adj -> StA -> cxW2
  half_t* A32a = (half_t*)alloc((long long)B * NN * 2);  // normbin  -> S_f16
  half_t* A32b = (half_t*)alloc((long long)B * NN * 2);  // adj_f16  -> StA_f16
  half_t* A32c = (half_t*)alloc((long long)B * NN * 2);  // St_f16   -> normc_f16
  half_t* A24  = (half_t*)alloc((long long)B * N * F * 2); // x_f16 -> cxW2_f16
  half_t* W1h  = (half_t*)alloc((long long)F * H * 2);
  half_t* W2h  = (half_t*)alloc((long long)H * F * 2);
  float*  xW1  = (float*)alloc((long long)B * N * H * 4);
  half_t* xW1h = (half_t*)alloc((long long)B * N * H * 2); // -> coarse_x_f16
  float*  x1   = (float*)alloc((long long)B * N * H * 4);
  half_t* x1h  = (half_t*)alloc((long long)B * N * H * 2);

  const dim3 blk(TB);
  const dim3 rowGrid(N, B);

  // Stage 0: f32 -> f16 operand conversions
  cvt_f32_to_f16<<<1024, blk, 0, stream>>>(x,   A24,  (long long)B * N * F);
  cvt_f32_to_f16<<<256,  blk, 0, stream>>>(W1,  W1h,  (long long)F * H);
  cvt_f32_to_f16<<<256,  blk, 0, stream>>>(W2,  W2h,  (long long)H * F);
  cvt_f32_to_f16<<<2048, blk, 0, stream>>>(adj, A32b, (long long)B * NN);

  // Stage 1: degree norms; materialize normbin(f16) + norm_adj(f32)
  row_stats_kernel<<<rowGrid, blk, 0, stream>>>(adj, dis_bin, dis_w, rowflag, N);
  norm_build_kernel<<<rowGrid, blk, 0, stream>>>(adj, dis_bin, dis_w, rowflag,
                                                 A32a, A64, N);

  // Stage 2: x1 = relu(normbin @ (x@W1) + b1)
  wmma_gemm_f16<<<dim3(H / 64, (B * N) / 128, 1), blk, 0, stream>>>(
      A24, W1h, xW1, B * N, F, H, 0, 0, 0);
  cvt_f32_to_f16<<<1024, blk, 0, stream>>>(xW1, xW1h, (long long)B * N * H);
  wmma_gemm_f16<<<dim3(H / 64, N / 128, B), blk, 0, stream>>>(
      A32a, xW1h, x1, N, N, H, NN, (long long)N * H, (long long)N * H);
  relu_bias_kernel<<<1024, blk, 0, stream>>>(x1, b1, x1h, H, (long long)B * N * H);

  // Stage 3: attention alpha, top-k cut, mask
  att_dot_kernel<<<B * N, blk, 0, stream>>>(x1, Watt, v_att, H);
  att_prop_kernel<<<rowGrid, blk, 0, stream>>>(adj, dis_bin, v_att, batt, alpha, N);
  topk_cut_kernel<<<B, blk, 0, stream>>>(alpha, nsnt, kptr, cutA, cutI, N);
  mask_cut_kernel<<<dim3((N + TB - 1) / TB, B), blk, 0, stream>>>(
      alpha, cutA, cutI, out + o_mask, cutalpha, N);

  // Stage 4: S (f32 out + f16 + transposed f16); normbin buffer becomes S16
  s_build_kernel<<<rowGrid, blk, 0, stream>>>(A64, cutalpha, out + o_S,
                                              A32a, A32c, N);

  // Stage 5: coarse_x = S^T x1 (direct to output), f16 copy, xs0
  wmma_gemm_f16<<<dim3(H / 64, N / 128, B), blk, 0, stream>>>(
      A32c, x1h, out + o_cx, N, N, H, NN, (long long)N * H, (long long)N * H);
  cvt_f32_to_f16<<<1024, blk, 0, stream>>>(out + o_cx, xW1h, (long long)B * N * H);
  row_mean_kernel<<<B * N, blk, 0, stream>>>(out + o_cx, out + o_xs0, H);

  // Stage 6: StA = S^T adj ; coarse_adj = StA S ; floor quantization
  wmma_gemm_f16<<<dim3(N / 64, N / 128, B), blk, 0, stream>>>(
      A32c, A32b, A64, N, N, N, NN, NN, NN);
  cvt_f32_to_f16<<<2048, blk, 0, stream>>>(A64, A32b, (long long)B * NN);
  wmma_gemm_f16<<<dim3(N / 64, N / 128, B), blk, 0, stream>>>(
      A32b, A32a, out + o_cadj, N, N, N, NN, NN, NN);
  quant_kernel<<<2048, blk, 0, stream>>>(out + o_cadj, (long long)B * NN);

  // Stage 7: coarse GCN: x2 = tanh(normc @ (coarse_x@W2) + b2); xs1
  coarse_stats_kernel<<<rowGrid, blk, 0, stream>>>(out + o_cadj, dis_c, N);
  normc_build_kernel<<<rowGrid, blk, 0, stream>>>(out + o_cadj, dis_c, A32c, N);
  wmma_gemm_f16<<<dim3(F / 64, (B * N) / 128, 1), blk, 0, stream>>>(
      xW1h, W2h, A64, B * N, H, F, 0, 0, 0);
  cvt_f32_to_f16<<<2048, blk, 0, stream>>>(A64, A24, (long long)B * N * F);
  wmma_gemm_f16<<<dim3(F / 64, N / 128, B), blk, 0, stream>>>(
      A32c, A24, out + o_x2, N, N, F, NN, (long long)N * F, (long long)N * F);
  tanh_bias_kernel<<<2048, blk, 0, stream>>>(out + o_x2, b2, F, (long long)B * N * F);
  row_mean_kernel<<<B * N, blk, 0, stream>>>(out + o_x2, out + o_xs1, F);
}
